// CoefficientNetwork_62380105007996
// MI455X (gfx1250) — compile-verified, matrix operand order checked by reading, not run on hardware
//
#include <hip/hip_runtime.h>
#include <hip/hip_bf16.h>

// ---------------------------------------------------------------------------
// CDNA5 (gfx1250) implementation of CoefficientNetwork.
// All heavy math is NT GEMM C = A * Bt^T with bf16 operands / f32 accum via
// v_wmma_f32_16x16x32_bf16. Tiles are staged in LDS with the gfx1250 async
// path (global_load_async_to_lds_b128 + s_wait_asynccnt), double-buffered.
// eigh-based inverse-sqrt is replaced by a Newton-Schulz coupled iteration
// (all-symmetric operands -> transpose-free inner loop).
// ---------------------------------------------------------------------------

typedef unsigned short u16;
typedef __attribute__((ext_vector_type(16))) __bf16 v16bf;
typedef __attribute__((ext_vector_type(8)))  float  v8f;
typedef __attribute__((ext_vector_type(4)))  float  vf4;

union Frag {
    v16bf v;
    vf4   q[2];
};

#define NDIM   2048
#define HID    256
#define HEADS  8
#define DH     32
#define LAYERS 4
#define NOCC   1024
#define NATOMS 512
#define N2     ((size_t)NDIM * NDIM)
#define NH     ((size_t)NDIM * HID)
#define NS_ITERS 14

// LDS tile: 128 rows x 32 K-elements, rows padded to 40 elems (80B) so the
// 16B ds_load_b128 fragment reads map to distinct banks (20*r mod 64 is a
// permutation over r=0..15).
#define TROW 40

__device__ __forceinline__ u16 f2bf(float x) {
    unsigned int u = __float_as_uint(x);
    unsigned int r = u + 0x7FFFu + ((u >> 16) & 1u);
    return (u16)(r >> 16);
}

// Issue one 128x32 bf16 tile (rows clamped into [0,limit)) into LDS via the
// async LDS-load path. 512 16B chunks, 256 threads -> 2 async ops per thread.
__device__ __forceinline__ void issue_tile_async(
    const u16* __restrict__ gbase, int ld, int row0, int limit, int kk,
    unsigned int lds_base, int tid)
{
#pragma unroll
    for (int t = 0; t < 2; ++t) {
        int chunk = tid + t * 256;
        int row = chunk >> 2;        // 0..127
        int c = chunk & 3;           // 16B chunk within the 64B row
        int grow = row0 + row;
        if (grow >= limit) grow = limit - 1;   // clamp: OOB tiles feed only
                                               // never-stored accumulators
        const u16* gp = gbase + (size_t)grow * ld + kk + c * 8;
        unsigned int laddr = lds_base + (unsigned int)(row * TROW + c * 8) * 2u;
        unsigned long long ga = (unsigned long long)(size_t)gp;
        asm volatile("global_load_async_to_lds_b128 %0, %1, off"
                     :: "v"(laddr), "v"(ga) : "memory");
    }
}

// ---------------------------------------------------------------------------
// WMMA GEMM (NT): C[M,N] = alpha * A[M,K](bf16) * Bt[N,K](bf16)^T  (+bias,
// +diag_add on the global diagonal, optional SiLU).
// Block: 256 threads = 8 wave32, tile 128x128, K-step 32, async double-buffer.
// Fragment layout per CDNA5 ISA 7.12.2 (16-bit A 16x32): lane&15 = row,
// lane>>4 selects K-halves {0..7,16..23} vs {8..15,24..31}.
// ---------------------------------------------------------------------------
__global__ __launch_bounds__(256) void wmma_gemm_nt(
    const u16* __restrict__ A, int lda,
    const u16* __restrict__ Bt, int ldb,
    float* __restrict__ C, int ldc,
    int M, int N, int K,
    float alpha, const float* __restrict__ bias, float diag_add, int act)
{
    __shared__ __align__(16) u16 tileA[2][128 * TROW];
    __shared__ __align__(16) u16 tileB[2][128 * TROW];

    const int tid  = threadIdx.x;
    const int lane = tid & 31;
    const int wave = tid >> 5;
    const int wm = wave & 3;          // 4 wave rows
    const int wn = wave >> 2;         // 2 wave cols
    const int bm = blockIdx.x * 128;
    const int bn = blockIdx.y * 128;
    const int r15 = lane & 15;
    const int sel = lane >> 4;

    const unsigned int ldsA0 = (unsigned int)(size_t)&tileA[0][0];
    const unsigned int ldsA1 = (unsigned int)(size_t)&tileA[1][0];
    const unsigned int ldsB0 = (unsigned int)(size_t)&tileB[0][0];
    const unsigned int ldsB1 = (unsigned int)(size_t)&tileB[1][0];

    v8f acc[2][4];
#pragma unroll
    for (int i = 0; i < 2; ++i)
#pragma unroll
        for (int j = 0; j < 4; ++j)
            acc[i][j] = (v8f)0.0f;

    const int nsteps = K >> 5;

    // Prologue: async-fill buffer 0.
    issue_tile_async(A,  lda, bm, M, 0, ldsA0, tid);
    issue_tile_async(Bt, ldb, bn, N, 0, ldsB0, tid);

    for (int step = 0; step < nsteps; ++step) {
        const int cur = step & 1;
        // Issue next K-step into the other buffer (freed by last barrier).
        if (step + 1 < nsteps) {
            int kk = (step + 1) << 5;
            issue_tile_async(A,  lda, bm, M, kk, cur ? ldsA0 : ldsA1, tid);
            issue_tile_async(Bt, ldb, bn, N, kk, cur ? ldsB0 : ldsB1, tid);
            asm volatile("s_wait_asynccnt 0x4" ::: "memory");  // cur batch done
        } else {
            asm volatile("s_wait_asynccnt 0x0" ::: "memory");
        }
        __syncthreads();   // all waves' async fills for `cur` visible

        const u16* lA = &tileA[cur][0];
        const u16* lB = &tileB[cur][0];
        Frag a[2], b[4];
#pragma unroll
        for (int i = 0; i < 2; ++i) {
            int row = wm * 32 + i * 16 + r15;
            a[i].q[0] = *reinterpret_cast<const vf4*>(lA + row * TROW + sel * 8);
            a[i].q[1] = *reinterpret_cast<const vf4*>(lA + row * TROW + 16 + sel * 8);
        }
#pragma unroll
        for (int j = 0; j < 4; ++j) {
            int row = wn * 64 + j * 16 + r15;
            b[j].q[0] = *reinterpret_cast<const vf4*>(lB + row * TROW + sel * 8);
            b[j].q[1] = *reinterpret_cast<const vf4*>(lB + row * TROW + 16 + sel * 8);
        }
#pragma unroll
        for (int i = 0; i < 2; ++i)
#pragma unroll
            for (int j = 0; j < 4; ++j)
                acc[i][j] = __builtin_amdgcn_wmma_f32_16x16x32_bf16(
                    false, a[i].v, false, b[j].v, (short)0, acc[i][j], false, false);

        __syncthreads();   // tile `cur` free for the next async fill
    }

    // Epilogue: C/D layout (ISA 7.12.2): lane&15 = column, VGPR r holds
    // row r + 8*(lane>>4).
#pragma unroll
    for (int i = 0; i < 2; ++i) {
        int rowb = bm + wm * 32 + i * 16 + sel * 8;
#pragma unroll
        for (int j = 0; j < 4; ++j) {
            int col = bn + wn * 64 + j * 16 + r15;
            if (col >= N) continue;
            float bv = bias ? bias[col] : 0.0f;
#pragma unroll
            for (int r = 0; r < 8; ++r) {
                int rg = rowb + r;
                if (rg >= M) break;
                float v = acc[i][j][r] * alpha + bv;
                if (rg == col) v += diag_add;
                if (act) v = v / (1.0f + __expf(-v));   // SiLU
                C[(size_t)rg * ldc + col] = v;
            }
        }
    }
}

// ---------------------------------------------------------------------------
// f32 -> bf16 convert (strided submatrix), optional device-scalar scaling.
// mode: 0 none, 1 *= rsqrt(*sp)  (Y0 = M/||M||_F), 2 *= (*sp)^(-1/4)
// (final M^{-1/2} = Zns * ||M||_F^{-1/2} folded into bf16 conversion).
// ---------------------------------------------------------------------------
__global__ void cvt_f32_bf16(const float* __restrict__ in, int ldi,
                             u16* __restrict__ out, int ldo,
                             int rows, int cols, int mode,
                             const float* __restrict__ sp)
{
    float f = 1.0f;
    if (mode == 1) f = rsqrtf(*sp);
    else if (mode == 2) f = rsqrtf(sqrtf(*sp));
    size_t n = (size_t)rows * cols;
    for (size_t idx = (size_t)blockIdx.x * blockDim.x + threadIdx.x;
         idx < n; idx += (size_t)gridDim.x * blockDim.x) {
        int r = (int)(idx / cols), c = (int)(idx % cols);
        out[(size_t)r * ldo + c] = f2bf(in[(size_t)r * ldi + c] * f);
    }
}

__global__ void scale_f32(const float* __restrict__ in, float* __restrict__ out,
                          size_t n, int mode, const float* __restrict__ sp)
{
    float f = 1.0f;
    if (mode == 1) f = rsqrtf(*sp);
    else if (mode == 2) f = rsqrtf(sqrtf(*sp));
    for (size_t i = (size_t)blockIdx.x * blockDim.x + threadIdx.x;
         i < n; i += (size_t)gridDim.x * blockDim.x)
        out[i] = in[i] * f;
}

// LDS-tiled transpose f32[rows,cols] -> bf16[cols,rows]
__global__ void transpose_f32_bf16(const float* __restrict__ in, int ldi,
                                   int rows, int cols,
                                   u16* __restrict__ out, int ldo)
{
    __shared__ float t[32][33];
    int bc = blockIdx.x * 32;   // col tile
    int br = blockIdx.y * 32;   // row tile
    for (int i = threadIdx.y; i < 32; i += 8) {
        int r = br + i, c = bc + threadIdx.x;
        t[i][threadIdx.x] = (r < rows && c < cols) ? in[(size_t)r * ldi + c] : 0.0f;
    }
    __syncthreads();
    for (int i = threadIdx.y; i < 32; i += 8) {
        int orow = bc + i, ocol = br + threadIdx.x;
        if (orow < cols && ocol < rows)
            out[(size_t)orow * ldo + ocol] = f2bf(t[threadIdx.x][i]);
    }
}

__global__ void sumsq_kernel(const float* __restrict__ a, size_t n,
                             float* __restrict__ out)
{
    __shared__ float red[256];
    float s = 0.0f;
    for (size_t i = (size_t)blockIdx.x * 256 + threadIdx.x; i < n;
         i += (size_t)gridDim.x * 256) {
        float v = a[i]; s += v * v;
    }
    red[threadIdx.x] = s; __syncthreads();
    for (int o = 128; o > 0; o >>= 1) {
        if (threadIdx.x < o) red[threadIdx.x] += red[threadIdx.x + o];
        __syncthreads();
    }
    if (threadIdx.x == 0) atomicAdd(out, red[0]);
}

// Row softmax with 3-channel pairwise bias: one block per row.
__global__ void softmax_bias_kernel(float* __restrict__ L,
                                    const float* __restrict__ hm,
                                    const float* __restrict__ sm,
                                    const float* __restrict__ P,
                                    const float* __restrict__ wb,
                                    int h, int n, float scale)
{
    __shared__ float red[256];
    int tid = threadIdx.x;
    size_t off = (size_t)blockIdx.x * n;
    float w0 = wb[h], w1 = wb[HEADS + h], w2 = wb[2 * HEADS + h];
    float mx = -3.4e38f;
    for (int j = tid; j < n; j += 256) {
        float v = L[off + j] * scale + hm[off + j] * w0 + sm[off + j] * w1 + P[off + j] * w2;
        L[off + j] = v;
        mx = fmaxf(mx, v);
    }
    red[tid] = mx; __syncthreads();
    for (int o = 128; o > 0; o >>= 1) {
        if (tid < o) red[tid] = fmaxf(red[tid], red[tid + o]);
        __syncthreads();
    }
    mx = red[0]; __syncthreads();
    float s = 0.0f;
    for (int j = tid; j < n; j += 256) {
        float e = __expf(L[off + j] - mx);
        L[off + j] = e; s += e;
    }
    red[tid] = s; __syncthreads();
    for (int o = 128; o > 0; o >>= 1) {
        if (tid < o) red[tid] += red[tid + o];
        __syncthreads();
    }
    float inv = 1.0f / red[0];
    for (int j = tid; j < n; j += 256) L[off + j] *= inv;
}

__global__ void gather_rows_kernel(const float* __restrict__ src, int cols,
                                   const int* __restrict__ idx,
                                   float* __restrict__ dst)
{
    int i = blockIdx.x;
    size_t s = (size_t)idx[i] * cols, d = (size_t)i * cols;
    for (int c = threadIdx.x; c < cols; c += blockDim.x) dst[d + c] = src[s + c];
}

__global__ void ew_add_kernel(float* __restrict__ a, const float* __restrict__ b,
                              size_t n)
{
    for (size_t i = (size_t)blockIdx.x * blockDim.x + threadIdx.x; i < n;
         i += (size_t)gridDim.x * blockDim.x)
        a[i] += b[i];
}

__global__ void set_identity_bf16(u16* __restrict__ out, int n)
{
    size_t total = (size_t)n * n;
    for (size_t i = (size_t)blockIdx.x * blockDim.x + threadIdx.x; i < total;
         i += (size_t)gridDim.x * blockDim.x) {
        int r = (int)(i / n), c = (int)(i % n);
        out[i] = (r == c) ? (u16)0x3F80 : (u16)0;
    }
}

// ---------------------------------------------------------------------------
// Host orchestration
// ---------------------------------------------------------------------------
extern "C" void kernel_launch(void* const* d_in, const int* in_sizes, int n_in,
                              void* d_out, int out_size, void* d_ws, size_t ws_size,
                              hipStream_t stream)
{
    (void)in_sizes; (void)n_in; (void)out_size; (void)ws_size;

    const float* x_in   = (const float*)d_in[0];
    const int*   tok    = (const int*)d_in[1];
    const int*   oidx   = (const int*)d_in[2];
    const float* hmat   = (const float*)d_in[3];
    const float* smat   = (const float*)d_in[4];
    const float* embedW = (const float*)d_in[5];
    enum { PW1=0, PB1, PW2, PB2, PWQ, PBQ, PWK, PBK, PWV, PBV, PWB, PWO, PBO, PWQK };
    const float* W[LAYERS][14];
    for (int l = 0; l < LAYERS; ++l)
        for (int k = 0; k < 14; ++k)
            W[l][k] = (const float*)d_in[6 + l * 14 + k];

    char* ws = (char*)d_ws;
    size_t off = 0;
    auto alloc = [&](size_t bytes) -> void* {
        off = (off + 255) & ~(size_t)255;
        void* p = ws + off;
        off += bytes;
        return p;
    };
    u16* sB   = (u16*)alloc(N2 * 2);
    u16* yB   = (u16*)alloc(N2 * 2);
    u16* zB   = (u16*)alloc(N2 * 2);
    u16* tB   = (u16*)alloc(N2 * 2);
    u16* zmB  = (u16*)alloc(N2 * 2);
    u16* ztB  = (u16*)alloc(N2 * 2);
    u16* wtB  = (u16*)alloc(N2 * 2);
    u16* aB   = (u16*)alloc(N2 * 2);
    u16* cB   = (u16*)alloc(N2 * 2);
    u16* zsB  = (u16*)alloc(N2 * 2);
    u16* xoB  = (u16*)alloc(NH * 2);
    u16* embB = (u16*)alloc(NH * 2);
    u16* t1B  = (u16*)alloc(NH * 2);
    u16* qB   = (u16*)alloc(NH * 2);
    u16* kB   = (u16*)alloc(NH * 2);
    u16* vtB  = (u16*)alloc(NH * 2);
    u16* oB   = (u16*)alloc(NH * 2);
    u16* q2B  = (u16*)alloc(NH * 2);
    u16* k2B  = (u16*)alloc(NH * 2);
    u16* w1T  = (u16*)alloc(HID * HID * 2);
    u16* w2T  = (u16*)alloc(HID * HID * 2);
    u16* wqT  = (u16*)alloc(HID * HID * 2);
    u16* wkT  = (u16*)alloc(HID * HID * 2);
    u16* wvT  = (u16*)alloc(HID * HID * 2);
    u16* woT  = (u16*)alloc(HID * HID * 2);
    u16* wqkT = (u16*)alloc((size_t)2 * HID * HID * 2);
    float* Lf    = (float*)alloc(N2 * 4);
    float* Tf    = (float*)alloc(N2 * 4);
    float* Y2f   = (float*)alloc(N2 * 4);
    float* Z2f   = (float*)alloc(N2 * 4);
    float* Wf    = (float*)alloc(N2 * 4);
    float* Mf    = (float*)alloc(N2 * 4);
    float* Zmatf = (float*)alloc(N2 * 4);
    float* Cf    = (float*)alloc(N2 * 4);
    float* Pf    = (float*)alloc(N2 * 4);
    float* xof   = (float*)alloc(NH * 4);
    float* embf  = (float*)alloc(NH * 4);
    float* xtmpf = (float*)alloc(NH * 4);
    float* qf    = (float*)alloc(NH * 4);
    float* kf    = (float*)alloc(NH * 4);
    float* vf    = (float*)alloc(NH * 4);
    float* of    = (float*)alloc(NH * 4);
    float* qkf   = (float*)alloc((size_t)NDIM * 2 * HID * 4);
    float* ssq   = (float*)alloc(4);

    auto gemm = [&](const u16* A, int lda, const u16* Bt, int ldb,
                    float* C, int ldc, int M, int N, int K,
                    float alpha, const float* bias, float diag, int act) {
        dim3 g((M + 127) / 128, (N + 127) / 128, 1);
        hipLaunchKernelGGL(wmma_gemm_nt, g, dim3(256), 0, stream,
                           A, lda, Bt, ldb, C, ldc, M, N, K, alpha, bias, diag, act);
    };
    auto cvt = [&](const float* in, int ldi, u16* out, int ldo,
                   int rows, int cols, int mode, const float* sp) {
        size_t n = (size_t)rows * cols;
        int blocks = (int)((n + 255) / 256);
        if (blocks > 16384) blocks = 16384;
        hipLaunchKernelGGL(cvt_f32_bf16, dim3(blocks), dim3(256), 0, stream,
                           in, ldi, out, ldo, rows, cols, mode, sp);
    };
    auto transp = [&](const float* in, int ldi, int rows, int cols,
                      u16* out, int ldo) {
        dim3 g((cols + 31) / 32, (rows + 31) / 32, 1);
        hipLaunchKernelGGL(transpose_f32_bf16, g, dim3(32, 8), 0, stream,
                           in, ldi, rows, cols, out, ldo);
    };

    // Newton-Schulz: given SPD M (f32), produce zsB = bf16(M^{-1/2}).
    auto newton_schulz = [&](const float* Min) {
        hipMemsetAsync(ssq, 0, 4, stream);
        hipLaunchKernelGGL(sumsq_kernel, dim3(2048), dim3(256), 0, stream, Min, N2, ssq);
        cvt(Min, NDIM, yB, NDIM, NDIM, NDIM, 1, ssq);       // Y0 = M / ||M||_F
        hipLaunchKernelGGL(set_identity_bf16, dim3(8192), dim3(256), 0, stream, zB, NDIM);
        for (int it = 0; it < NS_ITERS; ++it) {
            gemm(zB, NDIM, yB, NDIM, Tf, NDIM, NDIM, NDIM, NDIM, -0.5f, nullptr, 1.5f, 0);
            cvt(Tf, NDIM, tB, NDIM, NDIM, NDIM, 0, nullptr);
            gemm(tB, NDIM, zB, NDIM, Z2f, NDIM, NDIM, NDIM, NDIM, 1.0f, nullptr, 0.0f, 0); // Z' = T*Z
            gemm(yB, NDIM, tB, NDIM, Y2f, NDIM, NDIM, NDIM, NDIM, 1.0f, nullptr, 0.0f, 0); // Y' = Y*T
            cvt(Z2f, NDIM, zB, NDIM, NDIM, NDIM, 0, nullptr);
            cvt(Y2f, NDIM, yB, NDIM, NDIM, NDIM, 0, nullptr);
        }
        cvt(Z2f, NDIM, zsB, NDIM, NDIM, NDIM, 2, ssq);      // * ||M||_F^{-1/2}
    };

    // ---- setup ----
    hipLaunchKernelGGL(gather_rows_kernel, dim3(NDIM), dim3(256), 0, stream, x_in, HID, oidx, xof);
    hipLaunchKernelGGL(gather_rows_kernel, dim3(NDIM), dim3(256), 0, stream, embedW, HID, tok, embf);
    cvt(smat, NDIM, sB, NDIM, NDIM, NDIM, 0, nullptr);

    // C0 = orthonormalise(I, S) = S^{-1/2}
    newton_schulz(smat);
    {
        int blocks = (int)((N2 + 255) / 256); if (blocks > 16384) blocks = 16384;
        hipLaunchKernelGGL(scale_f32, dim3(blocks), dim3(256), 0, stream, Z2f, Cf, N2, 2, ssq);
    }
    cvt(Cf, NDIM, cB, NDIM, NDIM, NDIM, 0, nullptr);
    gemm(cB, NDIM, cB, NDIM, Pf, NDIM, NDIM, NDIM, NOCC, 2.0f, nullptr, 0.0f, 0);

    const float attn_scale = 0.1767766953f;   // 1/sqrt(DH)
    const float z_scale    = 0.0625f;         // 1/sqrt(HID)

    for (int l = 0; l < LAYERS; ++l) {
        transp(W[l][PW1], HID, HID, HID, w1T, HID);
        transp(W[l][PW2], HID, HID, HID, w2T, HID);
        transp(W[l][PWQ], HID, HID, HID, wqT, HID);
        transp(W[l][PWK], HID, HID, HID, wkT, HID);
        transp(W[l][PWV], HID, HID, HID, wvT, HID);
        transp(W[l][PWO], HID, HID, HID, woT, HID);
        transp(W[l][PWQK], 2 * HID, HID, 2 * HID, wqkT, HID);

        // emb = silu(emb@w1+b1)@w2 + b2
        cvt(embf, HID, embB, HID, NDIM, HID, 0, nullptr);
        gemm(embB, HID, w1T, HID, xtmpf, HID, NDIM, HID, HID, 1.0f, W[l][PB1], 0.0f, 1);
        cvt(xtmpf, HID, t1B, HID, NDIM, HID, 0, nullptr);
        gemm(t1B, HID, w2T, HID, embf, HID, NDIM, HID, HID, 1.0f, W[l][PB2], 0.0f, 0);

        // xo += emb
        {
            int blocks = (int)((NH + 255) / 256); if (blocks > 16384) blocks = 16384;
            hipLaunchKernelGGL(ew_add_kernel, dim3(blocks), dim3(256), 0, stream, xof, embf, NH);
        }
        cvt(xof, HID, xoB, HID, NDIM, HID, 0, nullptr);

        // q,k,v
        gemm(xoB, HID, wqT, HID, qf, HID, NDIM, HID, HID, 1.0f, W[l][PBQ], 0.0f, 0);
        gemm(xoB, HID, wkT, HID, kf, HID, NDIM, HID, HID, 1.0f, W[l][PBK], 0.0f, 0);
        gemm(xoB, HID, wvT, HID, vf, HID, NDIM, HID, HID, 1.0f, W[l][PBV], 0.0f, 0);
        cvt(qf, HID, qB, HID, NDIM, HID, 0, nullptr);
        cvt(kf, HID, kB, HID, NDIM, HID, 0, nullptr);
        transp(vf, HID, NDIM, HID, vtB, NDIM);   // vT: [HID, NDIM]

        for (int h = 0; h < HEADS; ++h) {
            gemm(qB + h * DH, HID, kB + h * DH, HID, Lf, NDIM,
                 NDIM, NDIM, DH, 1.0f, nullptr, 0.0f, 0);        // q_h k_h^T
            hipLaunchKernelGGL(softmax_bias_kernel, dim3(NDIM), dim3(256), 0, stream,
                               Lf, hmat, smat, Pf, W[l][PWB], h, NDIM, attn_scale);
            cvt(Lf, NDIM, aB, NDIM, NDIM, NDIM, 0, nullptr);
            gemm(aB, NDIM, vtB + (size_t)h * DH * NDIM, NDIM, of + h * DH, HID,
                 NDIM, DH, NDIM, 1.0f, nullptr, 0.0f, 0);        // o_h = A v_h
        }
        cvt(of, HID, oB, HID, NDIM, HID, 0, nullptr);
        gemm(oB, HID, woT, HID, xof, HID, NDIM, HID, HID, 1.0f, W[l][PBO], 0.0f, 1);
        cvt(xof, HID, xoB, HID, NDIM, HID, 0, nullptr);

        // qk = xo @ w_qk ; Z = q2 k2^T / sqrt(HID)
        gemm(xoB, HID, wqkT, HID, qkf, 2 * HID, NDIM, 2 * HID, HID, 1.0f, nullptr, 0.0f, 0);
        cvt(qkf,       2 * HID, q2B, HID, NDIM, HID, 0, nullptr);
        cvt(qkf + HID, 2 * HID, k2B, HID, NDIM, HID, 0, nullptr);
        gemm(q2B, HID, k2B, HID, Zmatf, NDIM, NDIM, NDIM, HID, z_scale, nullptr, 0.0f, 0);

        // M = Z^T S Z
        cvt(Zmatf, NDIM, zmB, NDIM, NDIM, NDIM, 0, nullptr);
        transp(Zmatf, NDIM, NDIM, NDIM, ztB, NDIM);                                 // Z^T
        gemm(sB, NDIM, ztB, NDIM, Wf, NDIM, NDIM, NDIM, NDIM, 1.0f, nullptr, 0.0f, 0); // W = S Z
        transp(Wf, NDIM, NDIM, NDIM, wtB, NDIM);                                    // W^T
        gemm(ztB, NDIM, wtB, NDIM, Mf, NDIM, NDIM, NDIM, NDIM, 1.0f, nullptr, 0.0f, 0); // Z^T W

        newton_schulz(Mf);

        // C = Z * M^{-1/2}
        gemm(zmB, NDIM, zsB, NDIM, Cf, NDIM, NDIM, NDIM, NDIM, 1.0f, nullptr, 0.0f, 0);
        cvt(Cf, NDIM, cB, NDIM, NDIM, NDIM, 0, nullptr);
        gemm(cB, NDIM, cB, NDIM, Pf, NDIM, NDIM, NDIM, NOCC, 2.0f, nullptr, 0.0f, 0);
    }

    hipMemcpyAsync(d_out, Cf, N2 * 4, hipMemcpyDeviceToDevice, stream);
}
